// LightGraphQA_5763846111243
// MI455X (gfx1250) — matrix operand order repaired and checked by reading, not compiled, not run on hardware
//
#include <hip/hip_runtime.h>
#include <stdint.h>

#define WG 256

typedef __attribute__((ext_vector_type(16))) __bf16 v16bf;
typedef __attribute__((ext_vector_type(8)))  float  v8f;

union Frag { v16bf v; uint4 q[2]; };

__device__ __forceinline__ unsigned short f2bf(float f) {
  unsigned int u = __float_as_uint(f);
  u += 0x7FFFu + ((u >> 16) & 1u);   // round-to-nearest-even bf16
  return (unsigned short)(u >> 16);
}

// ---- gfx1250 async global->LDS copy (ASYNCcnt path), with safe fallback ----
#if __has_builtin(__builtin_amdgcn_global_load_async_to_lds_b128)
#define ASYNC_COPY 1
typedef int v4i __attribute__((vector_size(16)));                 // matches builtin prototype
typedef __attribute__((address_space(1))) v4i gv4i;               // global (AS1)
typedef __attribute__((address_space(3))) v4i lv4i;               // LDS (AS3)
__device__ __forceinline__ void cp_b128(void* lds, const void* g) {
  __builtin_amdgcn_global_load_async_to_lds_b128((gv4i*)g, (lv4i*)lds, 0, 0);
}
__device__ __forceinline__ void wait_async() {
#if __has_builtin(__builtin_amdgcn_s_wait_asynccnt)
  __builtin_amdgcn_s_wait_asynccnt(0);
#else
  asm volatile("s_wait_asynccnt 0" ::: "memory");
#endif
}
#else
__device__ __forceinline__ void cp_b128(void* lds, const void* g) {
  *(uint4*)lds = *(const uint4*)g;   // global load -> VGPR -> ds store
}
__device__ __forceinline__ void wait_async() {}
#endif

// ---------------- f32 -> bf16 cast ----------------
__global__ __launch_bounds__(WG)
void k_cast(const float* __restrict__ in, unsigned short* __restrict__ out, int n) {
  int i = blockIdx.x * WG + threadIdx.x;
  if (i < n) out[i] = f2bf(in[i]);
}

// ---------------- f32 [K,N] -> bf16 transposed [N,K] ----------------
__global__ __launch_bounds__(WG)
void k_castT(const float* __restrict__ W, unsigned short* __restrict__ WT, int K, int N) {
  int i = blockIdx.x * WG + threadIdx.x;
  if (i >= K * N) return;
  int n = i / K, k = i - n * K;
  WT[(size_t)n * K + k] = f2bf(W[(size_t)k * N + n]);
}

// ---------------- bf16 WMMA GEMM, C = act(A @ WT^T + bias) ----------------
// A [M,K] bf16 row-major; WT [N,K] bf16 row-major (pre-transposed weights).
// Block tile 128x64, 8 waves, each wave owns 32x32 (2x2 accumulators, 4 WMMAs/K-step).
// Double-buffered LDS fed by async global->LDS b128 copies.
// M % 128 == 0, N % 64 == 0, K % 32 == 0 (true for all shapes here).
__global__ __launch_bounds__(WG)
void k_gemm(const unsigned short* __restrict__ A,
            const unsigned short* __restrict__ WT,
            const float* __restrict__ bias,
            float* __restrict__ outF,
            unsigned short* __restrict__ outB,
            int M, int N, int K, int relu)
{
  __shared__ __align__(16) unsigned short As[2][128][32];  // A tile,  K-major
  __shared__ __align__(16) unsigned short Bs[2][64][32];   // W^T tile, K-major

  const int tid  = threadIdx.x;
  const int lane = tid & 31;
  const int wv   = tid >> 5;        // 8 waves
  const int wrow = wv & 3;          // 4 row tiles of 32
  const int wcol = wv >> 2;         // 2 col tiles of 32
  const int m0 = blockIdx.y * 128;
  const int n0 = blockIdx.x * 64;

  // tile loaders: A 128x32 (16 elems/thread = 2 x b128), B 64x32 (8 elems/thread = 1 x b128)
  const int lr = tid >> 1;          // A row 0..127
  const int lc = (tid & 1) << 4;    // 0 | 16
  const int br = tid >> 2;          // B row (output col) 0..63
  const int bc = (tid & 3) << 3;    // 0,8,16,24

  v8f acc[2][2];
#pragma unroll
  for (int i = 0; i < 2; ++i)
#pragma unroll
    for (int j = 0; j < 2; ++j)
      acc[i][j] = (v8f){0.f,0.f,0.f,0.f,0.f,0.f,0.f,0.f};

  const int nk = K >> 5;

  // prologue: tile 0 into buffer 0
  {
    const unsigned short* ga = A  + (size_t)(m0 + lr) * K + lc;
    const unsigned short* gb = WT + (size_t)(n0 + br) * K + bc;
    cp_b128(&As[0][lr][lc],     ga);
    cp_b128(&As[0][lr][lc + 8], ga + 8);
    cp_b128(&Bs[0][br][bc],     gb);
  }

  for (int kt = 0; kt < nk; ++kt) {
    wait_async();
    __syncthreads();
    const int cur = kt & 1;

    if (kt + 1 < nk) {               // issue next tile's async copies (other buffer)
      const int k1 = (kt + 1) << 5;
      const unsigned short* ga = A  + (size_t)(m0 + lr) * K + k1 + lc;
      const unsigned short* gb = WT + (size_t)(n0 + br) * K + k1 + bc;
      cp_b128(&As[cur ^ 1][lr][lc],     ga);
      cp_b128(&As[cur ^ 1][lr][lc + 8], ga + 8);
      cp_b128(&Bs[cur ^ 1][br][bc],     gb);
    }

    // fragments: A 16x32 layout (lanes 0-15: K{0..7,16..23}; 16-31: K{8..15,24..31});
    //            B 32x16 layout (lanes 0-15: K{0..7,16..23}; 16-31: K{8..15,24..31}) — same as A
    Frag a[2], b[2];
    const int am = lane & 15;
    const int kh = (lane < 16) ? 0 : 8;
    const int kb = (lane < 16) ? 0 : 16;
#pragma unroll
    for (int ti = 0; ti < 2; ++ti) {
      const int m = wrow * 32 + ti * 16 + am;
      a[ti].q[0] = *(const uint4*)(&As[cur][m][kh]);
      a[ti].q[1] = *(const uint4*)(&As[cur][m][kh + 16]);
      const int n = wcol * 32 + ti * 16 + am;
      b[ti].q[0] = *(const uint4*)(&Bs[cur][n][kh]);
      b[ti].q[1] = *(const uint4*)(&Bs[cur][n][kh + 16]);
    }
#pragma unroll
    for (int ti = 0; ti < 2; ++ti)
#pragma unroll
      for (int tj = 0; tj < 2; ++tj)
        acc[ti][tj] = __builtin_amdgcn_wmma_f32_16x16x32_bf16(
            false, a[ti].v, false, b[tj].v, (short)0, acc[ti][tj], false, false);
  }

  // epilogue: C/D f32 layout — VGPR i: row i (lanes 0-15) / row 8+i (lanes 16-31), col = lane%16
  const int lcol = lane & 15;
  const int lrow = (lane < 16) ? 0 : 8;
#pragma unroll
  for (int ti = 0; ti < 2; ++ti) {
#pragma unroll
    for (int tj = 0; tj < 2; ++tj) {
      const int col = n0 + wcol * 32 + tj * 16 + lcol;
      const int mB  = m0 + wrow * 32 + ti * 16 + lrow;
      const float bv = bias ? bias[col] : 0.f;
#pragma unroll
      for (int i = 0; i < 8; ++i) {
        float v = acc[ti][tj][i] + bv;
        if (relu) v = fmaxf(v, 0.f);
        size_t idx = (size_t)(mB + i) * N + col;
        if (outF) outF[idx] = v;
        if (outB) outB[idx] = f2bf(v);
      }
    }
  }
}

// ---------------- h1 = relu(P[src] + Q[dst]) (br1 folded into P) ----------------
__global__ __launch_bounds__(WG)
void k_h1(const float* __restrict__ P, const float* __restrict__ Q,
          unsigned short* __restrict__ h1)
{
  int row = blockIdx.x;                 // b*992 + e
  int b = row / 992, e = row - b * 992;
  int s = e / 31, j = e - s * 31;
  int d = j + (j >= s);                 // receiver index
  const float* p = P + (size_t)(b * 32 + s) * 512;
  const float* q = Q + (size_t)(b * 32 + d) * 512;
  unsigned short* o = h1 + (size_t)row * 512;
  for (int k = threadIdx.x; k < 512; k += WG)
    o[k] = f2bf(fmaxf(p[k] + q[k], 0.f));
}

// ---------------- agg[b,n,:] = sum over edges with dst==n (closed-form, no atomics) --------
__global__ __launch_bounds__(128)
void k_agg(const float* __restrict__ eff, float* __restrict__ agg)
{
  int bn = blockIdx.x;                  // b*32 + n
  int b = bn >> 5, n = bn & 31;
  int c = threadIdx.x;                  // 0..127
  const float* base = eff + (size_t)b * 992 * 128;
  float s = 0.f;
#pragma unroll
  for (int i = 0; i < 32; ++i) {
    if (i == n) continue;
    int e = i * 31 + (n < i ? n : n - 1);   // the unique edge (i -> n)
    s += base[(size_t)e * 128 + c];
  }
  agg[(size_t)bn * 128 + c] = s;
}

// ---------------- updb = bf16(concat(ent, agg)) ----------------
__global__ __launch_bounds__(WG)
void k_updb(const float* __restrict__ ent, const float* __restrict__ agg,
            unsigned short* __restrict__ upd)
{
  int r = blockIdx.x;                   // 0..2047
  for (int k = threadIdx.x; k < 640; k += WG) {
    float v = (k < 512) ? ent[(size_t)r * 512 + k] : agg[(size_t)r * 128 + (k - 512)];
    upd[(size_t)r * 640 + k] = f2bf(v);
  }
}

// ---------------- location head: out[b,t,:] = (ent[b,:4,:] @ Wl + bl).reshape(8) --------
__global__ __launch_bounds__(WG)
void k_head(const float* __restrict__ ent, const float* __restrict__ Wl,
            const float* __restrict__ bl, float* __restrict__ out, int t)
{
  int gid = blockIdx.x * WG + threadIdx.x;
  if (gid >= 64 * 8) return;
  int b = gid >> 3, r = gid & 7, obj = r >> 1, c = r & 1;
  const float* row = ent + (size_t)(b * 32 + obj) * 512;
  float s = bl[c];
  for (int k = 0; k < 512; ++k) s += row[k] * Wl[k * 2 + c];
  out[((size_t)b * 16 + t) * 8 + r] = s;
}

// ================= host =================
extern "C" void kernel_launch(void* const* d_in, const int* in_sizes, int n_in,
                              void* d_out, int out_size, void* d_ws, size_t ws_size,
                              hipStream_t stream)
{
  (void)in_sizes; (void)n_in; (void)out_size; (void)ws_size;

  const float* entity = (const float*)d_in[0];
  // d_in[1]=Rs, d_in[2]=Rr unused: incidence is one-hot with a closed form
  const float* Wr1 = (const float*)d_in[3];
  const float* br1 = (const float*)d_in[4];
  const float* Wr2 = (const float*)d_in[5];
  const float* br2 = (const float*)d_in[6];
  const float* Wr3 = (const float*)d_in[7];
  const float* br3 = (const float*)d_in[8];
  const float* Wr4 = (const float*)d_in[9];
  const float* br4 = (const float*)d_in[10];
  const float* Wo1 = (const float*)d_in[11];
  const float* bo1 = (const float*)d_in[12];
  const float* Wo2 = (const float*)d_in[13];
  const float* bo2 = (const float*)d_in[14];
  const float* Wl  = (const float*)d_in[15];
  const float* bl  = (const float*)d_in[16];

  char* ws = (char*)d_ws;
  size_t off = 0;
  auto alloc = [&](size_t bytes) -> char* {
    char* p = ws + off;
    off = (off + bytes + 255) & ~(size_t)255;
    return p;
  };

  // bf16 TRANSPOSED weights [N,K] (cast once per launch; deterministic)
  unsigned short* wb_r1a = (unsigned short*)alloc((size_t)512 * 512 * 2);
  unsigned short* wb_r1b = (unsigned short*)alloc((size_t)512 * 512 * 2);
  unsigned short* wb_r2  = (unsigned short*)alloc((size_t)512 * 512 * 2);
  unsigned short* wb_r3  = (unsigned short*)alloc((size_t)128 * 512 * 2);
  unsigned short* wb_r4  = (unsigned short*)alloc((size_t)128 * 128 * 2);
  unsigned short* wb_o1  = (unsigned short*)alloc((size_t)512 * 640 * 2);
  unsigned short* wb_o2  = (unsigned short*)alloc((size_t)512 * 512 * 2);
  // activations
  unsigned short* entb = (unsigned short*)alloc((size_t)2048 * 512 * 2);
  float* P    = (float*)alloc((size_t)2048 * 512 * 4);
  float* Q    = (float*)alloc((size_t)2048 * 512 * 4);
  float* ent0 = (float*)alloc((size_t)2048 * 512 * 4);
  float* ent1 = (float*)alloc((size_t)2048 * 512 * 4);
  float* agg  = (float*)alloc((size_t)2048 * 128 * 4);
  unsigned short* updb = (unsigned short*)alloc((size_t)2048 * 640 * 2);
  unsigned short* hob  = (unsigned short*)alloc((size_t)2048 * 512 * 2);
  unsigned short* bufA = (unsigned short*)alloc((size_t)63488 * 512 * 2); // h1, then h3
  char*           bufB = alloc((size_t)63488 * 512 * 2);                  // h2 (bf16), then eff (f32)

  auto cast = [&](const float* in, unsigned short* out, int n) {
    hipLaunchKernelGGL(k_cast, dim3((n + WG - 1) / WG), dim3(WG), 0, stream, in, out, n);
  };
  auto castT = [&](const float* W, unsigned short* WT, int K, int N) {
    hipLaunchKernelGGL(k_castT, dim3((K * N + WG - 1) / WG), dim3(WG), 0, stream, W, WT, K, N);
  };
  auto gemm = [&](const unsigned short* A, const unsigned short* WT, const float* bias,
                  float* oF, unsigned short* oB, int M, int N, int K, int relu) {
    hipLaunchKernelGGL(k_gemm, dim3(N / 64, M / 128), dim3(WG), 0, stream,
                       A, WT, bias, oF, oB, M, N, K, relu);
  };

  castT(Wr1,             wb_r1a, 512, 512);   // Wr1 top half (src), transposed
  castT(Wr1 + 512 * 512, wb_r1b, 512, 512);   // Wr1 bottom half (dst), transposed
  castT(Wr2, wb_r2, 512, 512);
  castT(Wr3, wb_r3, 512, 128);
  castT(Wr4, wb_r4, 128, 128);
  castT(Wo1, wb_o1, 640, 512);
  castT(Wo2, wb_o2, 512, 512);

  float* outp    = (float*)d_out;            // [B,T,8]
  float* lastEnt = outp + 64 * 16 * 8;       // [B,N,D]
  const float* cur = entity;

  for (int t = 0; t < 16; ++t) {
    float* nxt = (t == 15) ? lastEnt : ((t & 1) ? ent1 : ent0);

    cast(cur, entb, 2048 * 512);
    // P = ent@Wr1a + br1 ; Q = ent@Wr1b   (factored incidence gather)
    gemm(entb, wb_r1a, br1, P, nullptr, 2048, 512, 512, 0);
    gemm(entb, wb_r1b, nullptr, Q, nullptr, 2048, 512, 512, 0);
    hipLaunchKernelGGL(k_h1, dim3(63488), dim3(WG), 0, stream, P, Q, bufA);
    // relational MLP tail on 63488 rows
    gemm(bufA, wb_r2, br2, nullptr, (unsigned short*)bufB, 63488, 512, 512, 1);
    gemm((unsigned short*)bufB, wb_r3, br3, nullptr, bufA, 63488, 128, 512, 1);
    gemm(bufA, wb_r4, br4, (float*)bufB, nullptr, 63488, 128, 128, 1);  // eff (f32)
    hipLaunchKernelGGL(k_agg, dim3(2048), dim3(128), 0, stream, (const float*)bufB, agg);
    // object model
    hipLaunchKernelGGL(k_updb, dim3(2048), dim3(WG), 0, stream, cur, agg, updb);
    gemm(updb, wb_o1, bo1, nullptr, hob, 2048, 512, 640, 1);
    gemm(hob, wb_o2, bo2, nxt, nullptr, 2048, 512, 512, 0);
    hipLaunchKernelGGL(k_head, dim3(2), dim3(WG), 0, stream, nxt, Wl, bl, outp, t);

    cur = nxt;
  }
}
